// DeepSTModel_90692529422957
// MI455X (gfx1250) — matrix operand order repaired and checked by reading, not compiled, not run on hardware
//
#include <hip/hip_runtime.h>
#include <hip/hip_bf16.h>

typedef __bf16 bf16;
typedef __attribute__((ext_vector_type(16))) __bf16 v16bf;
typedef __attribute__((ext_vector_type(8)))  float  v8f;

#define N_NODES 50000
#define G_DIM   3000
#define G_PAD   3008   // G rounded up to multiple of 32 (WMMA K step)
#define H_DIM   512
#define L_DIM   128
#define E_EDGES 300000

union FragAB {
    v16bf v;
    uint4 q[2];
};

// ---------------------------------------------------------------------------
// Elementwise / prep kernels
// ---------------------------------------------------------------------------

// Cast f32 [rows,K] row-major -> bf16 [rows,Kp] row-major, zero-padding K..Kp.
__global__ void cast_pad_rows(const float* __restrict__ in, bf16* __restrict__ out,
                              long rows, int K, int Kp) {
    long idx = (long)blockIdx.x * blockDim.x + threadIdx.x;
    long total = rows * (long)Kp;
    if (idx >= total) return;
    int  k = (int)(idx % Kp);
    long r = idx / Kp;
    float v = (k < K) ? in[r * (long)K + k] : 0.0f;
    out[idx] = (bf16)v;
}

// W is [K,Nc] f32 row-major; produce Wt = B^T as bf16 [NcPad, Kp], zero-padded.
__global__ void transpose_cast_pad(const float* __restrict__ W, bf16* __restrict__ Wt,
                                   int K, int Nc, int Kp, int NcPad) {
    long idx = (long)blockIdx.x * blockDim.x + threadIdx.x;
    long total = (long)NcPad * Kp;
    if (idx >= total) return;
    int k = (int)(idx % Kp);
    int n = (int)(idx / Kp);
    float v = (k < K && n < Nc) ? W[(long)k * Nc + n] : 0.0f;
    Wt[idx] = (bf16)v;
}

// Degree: deg = 1 (self loop) + sum of edge weights into dst, then rsqrt in place.
__global__ void deg_init(float* __restrict__ deg, long n) {
    long idx = (long)blockIdx.x * blockDim.x + threadIdx.x;
    if (idx < n) deg[idx] = 1.0f;
}
__global__ void deg_accum(const int* __restrict__ dst, const float* __restrict__ ew,
                          float* __restrict__ deg, long e) {
    long idx = (long)blockIdx.x * blockDim.x + threadIdx.x;
    if (idx >= e) return;
    __hip_atomic_fetch_add(&deg[dst[idx]], ew[idx], __ATOMIC_RELAXED,
                           __HIP_MEMORY_SCOPE_AGENT);
}
__global__ void deg_finish(float* __restrict__ deg, long n) {
    long idx = (long)blockIdx.x * blockDim.x + threadIdx.x;
    if (idx < n) deg[idx] = rsqrtf(deg[idx]);   // deg >= 1 always
}

// out[n,c] = hW[n,c]*dinv[n]^2 + bias[c]   (self-loop contribution + bias)
__global__ void gcn_self(const float* __restrict__ hW, const float* __restrict__ dinv,
                         const float* __restrict__ bias, float* __restrict__ out,
                         int C, long rows) {
    long idx = (long)blockIdx.x * blockDim.x + threadIdx.x;
    long total = rows * (long)C;
    if (idx >= total) return;
    int  c = (int)(idx % C);
    long n = idx / C;
    float di = dinv[n];
    out[idx] = hW[idx] * di * di + bias[c];
}

// Scatter-add messages, 4 channels per thread:
// out[dst, c..c+3] += hW[src, c..c+3] * dinv[src]*ew*dinv[dst]   (C % 4 == 0)
__global__ void gcn_edges4(const int* __restrict__ src, const int* __restrict__ dst,
                           const float* __restrict__ ew, const float* __restrict__ dinv,
                           const float* __restrict__ hW, float* __restrict__ out,
                           int C, long e) {
    const int  C4 = C >> 2;
    long idx = (long)blockIdx.x * blockDim.x + threadIdx.x;
    long total = e * (long)C4;
    if (idx >= total) return;
    int  c4 = (int)(idx % C4);
    long ei = idx / C4;
    int  s  = src[ei];
    int  d  = dst[ei];
    float norm = dinv[s] * ew[ei] * dinv[d];
    const float4 hv = *(const float4*)(hW + (long)s * C + c4 * 4);
    float* o = out + (long)d * C + c4 * 4;
    __hip_atomic_fetch_add(&o[0], hv.x * norm, __ATOMIC_RELAXED, __HIP_MEMORY_SCOPE_AGENT);
    __hip_atomic_fetch_add(&o[1], hv.y * norm, __ATOMIC_RELAXED, __HIP_MEMORY_SCOPE_AGENT);
    __hip_atomic_fetch_add(&o[2], hv.z * norm, __ATOMIC_RELAXED, __HIP_MEMORY_SCOPE_AGENT);
    __hip_atomic_fetch_add(&o[3], hv.w * norm, __ATOMIC_RELAXED, __HIP_MEMORY_SCOPE_AGENT);
}

// out[n, colOff+c] = bf16( act( in[n,c]*s[c]+t[c] ) ), ldo-strided bf16 store.
__global__ void scale_act_cast(const float* __restrict__ in, int C,
                               const float* __restrict__ s, const float* __restrict__ t,
                               int relu, bf16* __restrict__ out, int ldo, int colOff,
                               long rows) {
    long idx = (long)blockIdx.x * blockDim.x + threadIdx.x;
    long total = rows * (long)C;
    if (idx >= total) return;
    int  c = (int)(idx % C);
    long n = idx / C;
    float v = in[idx];
    if (s) v = v * s[c] + t[c];
    if (relu && v < 0.0f) v = 0.0f;
    out[n * (long)ldo + colOff + c] = (bf16)v;
}

// ---------------------------------------------------------------------------
// WMMA bf16 GEMM: C[M,Nc] = A[M,Kp] * B[Kp,Nc]  (B given as Bt[NcPad,Kp])
// Block = 256 threads (8 waves). Each wave owns a 32x64 output tile:
// 2 M-tiles x 4 N-tiles = 8 accumulators (8 WMMAs per 32-K step).
// Block covers 256 rows x 64 cols. Kp % 32 == 0, NcPad % 64 == 0.
//
// OOB A rows are CLAMPED to M-1 (not zeroed): an A fragment row M=lc only
// contributes to output row M=lc of the same tile, which the epilogue masks.
// This keeps the inner loop free of EXEC divergence.
// Fragment layouts per CDNA5 ISA 7.12.2 (16-bit A 16x32, B 32x16).
// ---------------------------------------------------------------------------
__global__ __launch_bounds__(256) void gemm_bf16_wmma(
    const bf16* __restrict__ A, long lda,
    const bf16* __restrict__ Bt, long ldb,
    int M, int Nc, int Kp,
    const float* __restrict__ bias, const float* __restrict__ scale,
    const float* __restrict__ shift, int relu,
    float* __restrict__ outF, long ldF,
    bf16* __restrict__ outB, long ldB)
{
    const int tid  = threadIdx.x;
    const int wave = tid >> 5;
    const int lane = tid & 31;
    const int hf   = lane >> 4;   // lane half (0: lanes 0-15, 1: lanes 16-31)
    const int lc   = lane & 15;

    const int tileM  = blockIdx.y * 256 + wave * 32;
    const int tileN0 = blockIdx.x * 64;

    // A rows held by this lane (two M-tiles), clamped into range.
    long r0 = (long)tileM + lc;        if (r0 > (long)M - 1) r0 = (long)M - 1;
    long r1 = (long)tileM + 16 + lc;   if (r1 > (long)M - 1) r1 = (long)M - 1;
    const bf16* Arow0 = A + r0 * lda;
    const bf16* Arow1 = A + r1 * lda;

    const bf16* Brow[4];
#pragma unroll
    for (int nt = 0; nt < 4; ++nt)
        Brow[nt] = Bt + (long)(tileN0 + nt * 16 + lc) * ldb;  // Bt padded: in bounds

    v8f acc[2][4] = {{v8f{}, v8f{}, v8f{}, v8f{}},
                     {v8f{}, v8f{}, v8f{}, v8f{}}};

    for (int k0 = 0; k0 < Kp; k0 += 32) {
        // A fragments: elems 0..7 -> K = k0+8*hf+e ; elems 8..15 -> K = k0+16+8*hf+e
        FragAB a0, a1;
        a0.q[0] = *(const uint4*)(Arow0 + k0 + 8 * hf);
        a0.q[1] = *(const uint4*)(Arow0 + k0 + 16 + 8 * hf);
        a1.q[0] = *(const uint4*)(Arow1 + k0 + 8 * hf);
        a1.q[1] = *(const uint4*)(Arow1 + k0 + 16 + 8 * hf);
        __builtin_prefetch(Arow0 + k0 + 64, 0, 1);   // global_prefetch_b8
        __builtin_prefetch(Arow1 + k0 + 64, 0, 1);

        // B fragments: lane holds column (tileN0+nt*16+lc); elems 0..7 -> K = k0+8*hf+e ; elems 8..15 -> K = k0+16+8*hf+e
        FragAB b[4];
        const int kb = k0 + 8 * hf;
#pragma unroll
        for (int nt = 0; nt < 4; ++nt) {
            b[nt].q[0] = *(const uint4*)(Brow[nt] + kb);
            b[nt].q[1] = *(const uint4*)(Brow[nt] + kb + 16);
        }
#pragma unroll
        for (int nt = 0; nt < 4; ++nt)
            acc[0][nt] = __builtin_amdgcn_wmma_f32_16x16x32_bf16(
                false, a0.v, false, b[nt].v, (short)0, acc[0][nt], false, false);
#pragma unroll
        for (int nt = 0; nt < 4; ++nt)
            acc[1][nt] = __builtin_amdgcn_wmma_f32_16x16x32_bf16(
                false, a1.v, false, b[nt].v, (short)0, acc[1][nt], false, false);
    }

    // Epilogue: C/D layout -> lane column = lc, VGPR r holds row = 8*hf + r
#pragma unroll
    for (int mt = 0; mt < 2; ++mt) {
#pragma unroll
        for (int nt = 0; nt < 4; ++nt) {
            const int c = tileN0 + nt * 16 + lc;
            if (c < Nc) {
                const float bv = bias  ? bias[c]  : 0.0f;
                const float sv = scale ? scale[c] : 1.0f;
                const float tv = shift ? shift[c] : 0.0f;
#pragma unroll
                for (int r = 0; r < 8; ++r) {
                    const long m = (long)tileM + mt * 16 + 8 * hf + r;
                    if (m < (long)M) {
                        float v = acc[mt][nt][r] + bv;
                        v = v * sv + tv;
                        if (relu && v < 0.0f) v = 0.0f;
                        if (outF) outF[m * ldF + c] = v;
                        if (outB) outB[m * ldB + c] = (bf16)v;
                    }
                }
            }
        }
    }
}

// ---------------------------------------------------------------------------
// Host-side helpers
// ---------------------------------------------------------------------------
static inline unsigned blocks1d(long total) {
    return (unsigned)((total + 255) / 256);
}

static void run_gemm(hipStream_t s, const bf16* A, long lda, const bf16* Bt, long ldb,
                     int M, int Nc, int Kp,
                     const float* bias, const float* scale, const float* shift, int relu,
                     float* outF, long ldF, bf16* outB, long ldB) {
    dim3 grid((Nc + 63) / 64, (M + 255) / 256);
    gemm_bf16_wmma<<<grid, 256, 0, s>>>(A, lda, Bt, ldb, M, Nc, Kp,
                                        bias, scale, shift, relu,
                                        outF, ldF, outB, ldB);
}

static void run_transpose(hipStream_t s, const float* W, bf16* Wt,
                          int K, int Nc, int Kp, int NcPad) {
    transpose_cast_pad<<<blocks1d((long)NcPad * Kp), 256, 0, s>>>(W, Wt, K, Nc, Kp, NcPad);
}

extern "C" void kernel_launch(void* const* d_in, const int* in_sizes, int n_in,
                              void* d_out, int out_size, void* d_ws, size_t ws_size,
                              hipStream_t stream) {
    (void)in_sizes; (void)n_in; (void)out_size; (void)ws_size;

    const float* x      = (const float*)d_in[0];
    const int*   ei     = (const int*)d_in[1];
    const float* ew     = (const float*)d_in[2];
    const float* gcn_w1 = (const float*)d_in[3];
    const float* gcn_b1 = (const float*)d_in[4];
    const float* bn0_s  = (const float*)d_in[5];
    const float* bn0_t  = (const float*)d_in[6];
    const float* gcn_w2 = (const float*)d_in[7];
    const float* gcn_b2 = (const float*)d_in[8];
    const float* enc_w1 = (const float*)d_in[9];
    const float* enc_b1 = (const float*)d_in[10];
    const float* enc_s1 = (const float*)d_in[11];
    const float* enc_t1 = (const float*)d_in[12];
    const float* enc_w2 = (const float*)d_in[13];
    const float* enc_b2 = (const float*)d_in[14];
    const float* enc_s2 = (const float*)d_in[15];
    const float* enc_t2 = (const float*)d_in[16];
    const float* enc_w3 = (const float*)d_in[17];
    const float* enc_b3 = (const float*)d_in[18];
    const float* enc_s3 = (const float*)d_in[19];
    const float* enc_t3 = (const float*)d_in[20];
    const float* enc_w4 = (const float*)d_in[21];
    const float* enc_b4 = (const float*)d_in[22];
    const float* dec_w1 = (const float*)d_in[23];
    const float* dec_b1 = (const float*)d_in[24];
    const float* dec_s1 = (const float*)d_in[25];
    const float* dec_t1 = (const float*)d_in[26];
    const float* dec_w2 = (const float*)d_in[27];
    const float* dec_b2 = (const float*)d_in[28];
    const float* dec_s2 = (const float*)d_in[29];
    const float* dec_t2 = (const float*)d_in[30];
    const float* dec_w3 = (const float*)d_in[31];
    const float* dec_b3 = (const float*)d_in[32];
    const float* dec_s3 = (const float*)d_in[33];
    const float* dec_t3 = (const float*)d_in[34];
    const float* dec_w4 = (const float*)d_in[35];
    const float* dec_b4 = (const float*)d_in[36];
    const float* fus_w  = (const float*)d_in[37];
    const float* fus_b  = (const float*)d_in[38];
    const float* fus_s  = (const float*)d_in[39];
    const float* fus_t  = (const float*)d_in[40];

    const int* srcI = ei;             // edge_index[0]
    const int* dstI = ei + E_EDGES;   // edge_index[1]

    // ---- workspace carve (256B aligned chunks), total ~690 MB ----
    char* base = (char*)d_ws;
    size_t off = 0;
    auto carve = [&](size_t bytes) -> void* {
        void* p = base + off;
        off += (bytes + 255) & ~(size_t)255;
        return p;
    };

    bf16*  xB    = (bf16*)carve((size_t)N_NODES * G_PAD * 2);   // x as bf16, K padded
    bf16*  wtG1  = (bf16*)carve((size_t)512  * G_PAD * 2);
    bf16*  wtG2  = (bf16*)carve((size_t)128  * 512   * 2);
    bf16*  wtE1  = (bf16*)carve((size_t)512  * G_PAD * 2);
    bf16*  wtE2  = (bf16*)carve((size_t)256  * 512   * 2);
    bf16*  wtE3  = (bf16*)carve((size_t)128  * 256   * 2);
    bf16*  wtE4  = (bf16*)carve((size_t)128  * 128   * 2);
    bf16*  wtD1  = (bf16*)carve((size_t)128  * 128   * 2);
    bf16*  wtD2  = (bf16*)carve((size_t)256  * 128   * 2);
    bf16*  wtD3  = (bf16*)carve((size_t)512  * 256   * 2);
    bf16*  wtD4  = (bf16*)carve((size_t)3008 * 512   * 2);
    bf16*  wtF   = (bf16*)carve((size_t)128  * 256   * 2);
    float* dinv  = (float*)carve((size_t)N_NODES * 4);
    float* F1    = (float*)carve((size_t)N_NODES * 512 * 4);    // pre-agg GEMM out
    float* F2    = (float*)carve((size_t)N_NODES * 512 * 4);    // aggregated
    bf16*  gB    = (bf16*)carve((size_t)N_NODES * 512 * 2);     // BN+relu(g) bf16
    bf16*  h1B   = (bf16*)carve((size_t)N_NODES * 512 * 2);
    bf16*  h2B   = (bf16*)carve((size_t)N_NODES * 256 * 2);
    bf16*  h3B   = (bf16*)carve((size_t)N_NODES * 128 * 2);
    bf16*  combB = (bf16*)carve((size_t)N_NODES * 256 * 2);     // [gnn | dae] bf16
    // decoder buffers alias encoder buffers of equal size (lifetimes disjoint)
    bf16* d1B = h3B;
    bf16* d2B = h2B;
    bf16* d3B = h1B;

    float* outFused = (float*)d_out;                            // [N, L]
    float* outRecon = (float*)d_out + (size_t)N_NODES * L_DIM;  // [N, G]

    // ---- operand prep ----
    cast_pad_rows<<<blocks1d((long)N_NODES * G_PAD), 256, 0, stream>>>(x, xB, N_NODES, G_DIM, G_PAD);
    run_transpose(stream, gcn_w1, wtG1, G_DIM, H_DIM, G_PAD, 512);
    run_transpose(stream, gcn_w2, wtG2, 512, 128, 512, 128);
    run_transpose(stream, enc_w1, wtE1, G_DIM, 512, G_PAD, 512);
    run_transpose(stream, enc_w2, wtE2, 512, 256, 512, 256);
    run_transpose(stream, enc_w3, wtE3, 256, 128, 256, 128);
    run_transpose(stream, enc_w4, wtE4, 128, 128, 128, 128);
    run_transpose(stream, dec_w1, wtD1, 128, 128, 128, 128);
    run_transpose(stream, dec_w2, wtD2, 128, 256, 128, 256);
    run_transpose(stream, dec_w3, wtD3, 256, 512, 256, 512);
    run_transpose(stream, dec_w4, wtD4, 512, G_DIM, 512, 3008);
    run_transpose(stream, fus_w,  wtF,  256, 128, 256, 128);

    // ---- degree / symmetric normalization ----
    deg_init  <<<blocks1d(N_NODES), 256, 0, stream>>>(dinv, N_NODES);
    deg_accum <<<blocks1d(E_EDGES), 256, 0, stream>>>(dstI, ew, dinv, E_EDGES);
    deg_finish<<<blocks1d(N_NODES), 256, 0, stream>>>(dinv, N_NODES);

    // ---- GCN layer 1: hW = x@W1 ; aggregate ; BN+relu -> gB ----
    run_gemm(stream, xB, G_PAD, wtG1, G_PAD, N_NODES, H_DIM, G_PAD,
             nullptr, nullptr, nullptr, 0, F1, H_DIM, nullptr, 0);
    gcn_self  <<<blocks1d((long)N_NODES * H_DIM),     256, 0, stream>>>(F1, dinv, gcn_b1, F2, H_DIM, N_NODES);
    gcn_edges4<<<blocks1d((long)E_EDGES * (H_DIM/4)), 256, 0, stream>>>(srcI, dstI, ew, dinv, F1, F2, H_DIM, E_EDGES);
    scale_act_cast<<<blocks1d((long)N_NODES * H_DIM), 256, 0, stream>>>(F2, H_DIM, bn0_s, bn0_t, 1, gB, H_DIM, 0, N_NODES);

    // ---- GCN layer 2: hW = g@W2 ; aggregate -> gnn_latent -> combined[:, :128] ----
    run_gemm(stream, gB, H_DIM, wtG2, H_DIM, N_NODES, L_DIM, H_DIM,
             nullptr, nullptr, nullptr, 0, F1, L_DIM, nullptr, 0);
    gcn_self  <<<blocks1d((long)N_NODES * L_DIM),     256, 0, stream>>>(F1, dinv, gcn_b2, F2, L_DIM, N_NODES);
    gcn_edges4<<<blocks1d((long)E_EDGES * (L_DIM/4)), 256, 0, stream>>>(srcI, dstI, ew, dinv, F1, F2, L_DIM, E_EDGES);
    scale_act_cast<<<blocks1d((long)N_NODES * L_DIM), 256, 0, stream>>>(F2, L_DIM, nullptr, nullptr, 0, combB, 2 * L_DIM, 0, N_NODES);

    // ---- DAE encoder (fused bias/scale/shift/relu epilogues, bf16 chaining) ----
    run_gemm(stream, xB, G_PAD, wtE1, G_PAD, N_NODES, 512, G_PAD,
             enc_b1, enc_s1, enc_t1, 1, nullptr, 0, h1B, 512);
    run_gemm(stream, h1B, 512, wtE2, 512, N_NODES, 256, 512,
             enc_b2, enc_s2, enc_t2, 1, nullptr, 0, h2B, 256);
    run_gemm(stream, h2B, 256, wtE3, 256, N_NODES, 128, 256,
             enc_b3, enc_s3, enc_t3, 1, nullptr, 0, h3B, 128);
    // dae_latent -> combined[:, 128:256] (bias only, no activation)
    run_gemm(stream, h3B, 128, wtE4, 128, N_NODES, 128, 128,
             enc_b4, nullptr, nullptr, 0, nullptr, 0, combB + 128, 256);

    // ---- DAE decoder (reads dae_latent in place from combined, lda=256) ----
    run_gemm(stream, combB + 128, 256, wtD1, 128, N_NODES, 128, 128,
             dec_b1, dec_s1, dec_t1, 1, nullptr, 0, d1B, 128);
    run_gemm(stream, d1B, 128, wtD2, 128, N_NODES, 256, 128,
             dec_b2, dec_s2, dec_t2, 1, nullptr, 0, d2B, 256);
    run_gemm(stream, d2B, 256, wtD3, 256, N_NODES, 512, 256,
             dec_b3, dec_s3, dec_t3, 1, nullptr, 0, d3B, 512);
    run_gemm(stream, d3B, 512, wtD4, 512, N_NODES, G_DIM, 512,
             dec_b4, nullptr, nullptr, 0, outRecon, G_DIM, nullptr, 0);

    // ---- fusion head: relu((combined@fus_w + b)*s + t) -> d_out[:, :L] ----
    run_gemm(stream, combB, 256, wtF, 256, N_NODES, L_DIM, 256,
             fus_b, fus_s, fus_t, 1, outFused, L_DIM, nullptr, 0);
}
